// MRCGNN_sub_77403900608993
// MI455X (gfx1250) — compile-verified, matrix operand order checked by reading, not run on hardware
//
#include <hip/hip_runtime.h>
#include <hip/hip_bf16.h>
#include <math.h>

#define NNODES 20000
#define NEDGES 320000

typedef __attribute__((ext_vector_type(16))) __bf16 v16bf;
typedef __attribute__((ext_vector_type(8)))  float  v8f;

// ---------------------------------------------------------------------------
// WMMA GEMM: C[M,Nc] = act(A[M,K] @ B[K,Nc] + bias),  B leading dim = LDB
// (compile-time so B loads use constant immediate offsets, no address math).
// One wave32 computes a 16x32 tile of C (two v_wmma_f32_16x16x32_bf16 per
// K-step sharing the A fragment; K unrolled 2x -> 4 static WMMAs).
// Fragment layouts per CDNA5 ISA 7.12.2:
//   A (16x32 bf16): lane l holds row M=l&15; element e -> K = 16*(e>>3) + 8*(l>>4) + (e&7)
//   B (32x16 bf16): lane l holds col N=l&15; same K mapping
//   C/D (16x16 f32): VGPR i -> M = i + 8*(l>>4), N = l&15
// ---------------------------------------------------------------------------
template <int LDB>
__global__ __launch_bounds__(128)
void hgt_wmma_gemm_bf16(const float* __restrict__ A, int lda,
                        const float* __restrict__ B,
                        const float* __restrict__ bias,
                        float* __restrict__ C, int ldc,
                        int M, int Nc, int K, int act)
{
  const int wave   = blockIdx.x * (blockDim.x >> 5) + (threadIdx.x >> 5);
  const int tilesN = Nc >> 5;                 // 32-wide N tiles
  const int tiles  = (M >> 4) * tilesN;
  if (wave >= tiles) return;
  const int row0 = (wave / tilesN) << 4;
  const int col0 = (wave % tilesN) << 5;

  const int lane = threadIdx.x & 31;
  const int lo   = lane & 15;
  const int hi   = lane >> 4;
  const int kb   = hi << 3;                   // per-lane K sub-offset

  v8f acc0 = {0.f,0.f,0.f,0.f,0.f,0.f,0.f,0.f};
  v8f acc1 = {0.f,0.f,0.f,0.f,0.f,0.f,0.f,0.f};

  const float* Arow  = A + (size_t)(row0 + lo) * lda;
  // base for this lane's B column/K-suboffset; advanced by constant 32*LDB
  const float* Bbase = B + (size_t)kb * LDB + col0 + lo;

#pragma unroll 2
  for (int k0 = 0; k0 < K; k0 += 32) {
    // speculative prefetch of next K-step of this lane's A row
    __builtin_prefetch(Arow + k0 + 32 + kb, 0, 1);

    // A fragment: two aligned float4 pairs (K = k0+kb+0..7 and k0+16+kb+0..7)
    const float4 a0 = *(const float4*)(Arow + k0 + kb);
    const float4 a1 = *(const float4*)(Arow + k0 + kb + 4);
    const float4 a2 = *(const float4*)(Arow + k0 + 16 + kb);
    const float4 a3 = *(const float4*)(Arow + k0 + 16 + kb + 4);
    const float af[16] = {a0.x,a0.y,a0.z,a0.w, a1.x,a1.y,a1.z,a1.w,
                          a2.x,a2.y,a2.z,a2.w, a3.x,a3.y,a3.z,a3.w};
    v16bf a, b0, b1;
#pragma unroll
    for (int e = 0; e < 16; ++e) a[e] = (__bf16)af[e];

    // B fragments: constant immediate offsets off one base pointer
#pragma unroll
    for (int e = 0; e < 8; ++e) {
      b0[e]     = (__bf16)Bbase[e * LDB];
      b1[e]     = (__bf16)Bbase[e * LDB + 16];
      b0[e + 8] = (__bf16)Bbase[(e + 16) * LDB];
      b1[e + 8] = (__bf16)Bbase[(e + 16) * LDB + 16];
    }
    Bbase += 32 * LDB;

    acc0 = __builtin_amdgcn_wmma_f32_16x16x32_bf16(false, a, false, b0,
                                                   (short)0, acc0, false, false);
    acc1 = __builtin_amdgcn_wmma_f32_16x16x32_bf16(false, a, false, b1,
                                                   (short)0, acc1, false, false);
  }

  // epilogue: bias hoisted per lane, strength-reduced C addressing
  const float bv0 = bias ? bias[col0 + lo]      : 0.f;
  const float bv1 = bias ? bias[col0 + 16 + lo] : 0.f;
  float* Crow = C + (size_t)(row0 + kb) * ldc + col0 + lo;
#pragma unroll
  for (int i = 0; i < 8; ++i) {
    float v0 = acc0[i] + bv0;
    float v1 = acc1[i] + bv1;
    if (act == 1) { v0 = fmaxf(v0, 0.f); v1 = fmaxf(v1, 0.f); }
    Crow[0]  = v0;
    Crow[16] = v1;
    Crow += ldc;
  }
}

// ------------------- order-preserving float <-> uint mapping ----------------
__device__ __forceinline__ unsigned okey(float f) {
  unsigned u = __float_as_uint(f);
  return (u & 0x80000000u) ? ~u : (u | 0x80000000u);
}
__device__ __forceinline__ float oval(unsigned k) {
  unsigned u = (k & 0x80000000u) ? (k & 0x7fffffffu) : ~k;
  return __uint_as_float(u);
}

// ------------------- edge attention logits: one wave per edge ---------------
__global__ void hgt_edge_logits(const float* __restrict__ q,
                                const float* __restrict__ k2,
                                const int* __restrict__ src,
                                const int* __restrict__ dst,
                                const float* __restrict__ p_rel,
                                float* __restrict__ alpha,
                                int E, int H, int D)
{
  const int e = blockIdx.x * (blockDim.x >> 5) + (threadIdx.x >> 5);
  if (e >= E) return;
  const int lane = threadIdx.x & 31;
  const int s = src[e], d = dst[e];
  const float inv = 1.0f / sqrtf((float)D);
  for (int h = 0; h < H; ++h) {
    const float* qp = q  + (size_t)d * H * D + h * D;
    const float* kp = k2 + (size_t)s * H * D + h * D;
    float acc = 0.f;
    for (int i = lane; i < D; i += 32) acc += qp[i] * kp[i];
#pragma unroll
    for (int off = 16; off > 0; off >>= 1) acc += __shfl_down(acc, off, 32);
    if (lane == 0) alpha[(size_t)e * H + h] = acc * p_rel[h] * inv;
  }
}

__global__ void hgt_fill_u32(unsigned* __restrict__ p, unsigned v, int n) {
  int i = blockIdx.x * blockDim.x + threadIdx.x;
  if (i < n) p[i] = v;
}
__global__ void hgt_zero_f32(float* __restrict__ p, long long n) {
  long long i = (long long)blockIdx.x * blockDim.x + threadIdx.x;
  if (i < n) p[i] = 0.f;
}

__global__ void hgt_seg_max(const float* __restrict__ alpha,
                            const int* __restrict__ dst,
                            unsigned* __restrict__ amaxk, int E, int H)
{
  int i = blockIdx.x * blockDim.x + threadIdx.x;
  if (i >= E * H) return;
  int e = i / H, h = i - e * H;
  atomicMax(&amaxk[(size_t)dst[e] * H + h], okey(alpha[i]));
}

// in-place alpha -> exp(alpha - amax[dst]); accumulate denominator
__global__ void hgt_exp_den(float* __restrict__ alpha,
                            const int* __restrict__ dst,
                            const unsigned* __restrict__ amaxk,
                            float* __restrict__ den, int E, int H)
{
  int i = blockIdx.x * blockDim.x + threadIdx.x;
  if (i >= E * H) return;
  int e = i / H, h = i - e * H;
  float am = oval(amaxk[(size_t)dst[e] * H + h]);
  if (!__builtin_isfinite(am)) am = 0.f;
  float ex = expf(alpha[i] - am);
  alpha[i] = ex;
  atomicAdd(&den[(size_t)dst[e] * H + h], ex);
}

// scatter v2[src] * attn into out[dst]: one wave per edge, lanes over D
__global__ void hgt_scatter_v(const float* __restrict__ ex,
                              const float* __restrict__ v2,
                              const int* __restrict__ src,
                              const int* __restrict__ dst,
                              const float* __restrict__ den,
                              float* __restrict__ outp,
                              int E, int H, int D)
{
  const int e = blockIdx.x * (blockDim.x >> 5) + (threadIdx.x >> 5);
  if (e >= E) return;
  const int lane = threadIdx.x & 31;
  const int s = src[e], d = dst[e];
  for (int h = 0; h < H; ++h) {
    const float attn = ex[(size_t)e * H + h] /
                       fmaxf(den[(size_t)d * H + h], 1e-16f);
    const float* vp = v2   + (size_t)s * H * D + h * D;
    float*       op = outp + (size_t)d * H * D + h * D;
    for (int i = lane; i < D; i += 32) atomicAdd(&op[i], vp[i] * attn);
  }
}

// exact (erf) gelu, in place
__global__ void hgt_gelu(float* __restrict__ p, long long n) {
  long long i = (long long)blockIdx.x * blockDim.x + threadIdx.x;
  if (i < n) {
    float x = p[i];
    p[i] = 0.5f * x * (1.0f + erff(x * 0.70710678118654752440f));
  }
}

// per-column mean: block per column
__global__ void hgt_col_mean(const float* __restrict__ X, float* __restrict__ outp,
                             int n, int cols)
{
  __shared__ float sm[256];
  const int c = blockIdx.x;
  float acc = 0.f;
  for (int r = threadIdx.x; r < n; r += blockDim.x)
    acc += X[(size_t)r * cols + c];
  sm[threadIdx.x] = acc;
  __syncthreads();
  for (int s = blockDim.x >> 1; s > 0; s >>= 1) {
    if ((int)threadIdx.x < s) sm[threadIdx.x] += sm[threadIdx.x + s];
    __syncthreads();
  }
  if (threadIdx.x == 0) outp[c] = sm[0] / (float)n;
}

// ---------------------------------------------------------------------------
extern "C" void kernel_launch(void* const* d_in, const int* in_sizes, int n_in,
                              void* d_out, int out_size, void* d_ws, size_t ws_size,
                              hipStream_t stream) {
  (void)in_sizes; (void)n_in; (void)out_size; (void)ws_size;
  const int N = NNODES, E = NEDGES, H = 2;

  const float* x         = (const float*)d_in[0];
  const float* x_sim     = (const float*)d_in[1];
  const int*   edge      = (const int*)d_in[2];
  const int*   src       = edge;
  const int*   dst       = edge + E;
  const float* lin_W     = (const float*)d_in[3];
  const float* lin_b     = (const float*)d_in[4];
  const float* lin_sim_W = (const float*)d_in[5];
  const float* lin_sim_b = (const float*)d_in[6];

  struct EncP { const float *Wk,*bk,*Wq,*bq,*Wv,*bv,*Wa,*ba,*a_rel,*m_rel,*p_rel; int fin, fout; };
  auto getEnc = [&](int base, int fin, int fout) {
    EncP p;
    p.Wk = (const float*)d_in[base+0];  p.bk = (const float*)d_in[base+1];
    p.Wq = (const float*)d_in[base+2];  p.bq = (const float*)d_in[base+3];
    p.Wv = (const float*)d_in[base+4];  p.bv = (const float*)d_in[base+5];
    p.Wa = (const float*)d_in[base+6];  p.ba = (const float*)d_in[base+7];
    p.a_rel = (const float*)d_in[base+8];
    p.m_rel = (const float*)d_in[base+9];
    p.p_rel = (const float*)d_in[base+10];
    p.fin = fin; p.fout = fout;
    return p;
  };
  EncP enc1  = getEnc(7,  512, 256);
  EncP enc2  = getEnc(18, 256, 128);
  EncP encs1 = getEnc(29, 256, 256);
  EncP encs2 = getEnc(40, 256, 128);

  // ---- workspace partition (floats) ----
  float* ws = (float*)d_ws;
  size_t o = 0;
  float* act_in = ws + o; o += (size_t)N * 512;
  float* kbuf   = ws + o; o += (size_t)N * 256;
  float* qbuf   = ws + o; o += (size_t)N * 256;
  float* vbuf   = ws + o; o += (size_t)N * 256;
  float* k2buf  = ws + o; o += (size_t)N * 256;
  float* v2buf  = ws + o; o += (size_t)N * 256;
  float* aggbuf = ws + o; o += (size_t)N * 256;
  float* alphaB = ws + o; o += (size_t)E * H;
  float* den    = ws + o; o += (size_t)N * H;
  unsigned* amaxk = (unsigned*)(ws + o); o += (size_t)N * H;

  // ---- output partition ----
  float* out      = (float*)d_out;
  float* h_os     = out;
  float* h_os_sim = out + 128;
  float* x1_o     = out + 256;
  float* x2_o     = x1_o + (size_t)N * 256;
  float* x1_s     = x2_o + (size_t)N * 128;
  float* x2_s     = x1_s + (size_t)N * 256;

  auto gemm = [&](const float* A, int lda, const float* B, int ldb,
                  const float* bias, float* C, int ldc,
                  int M, int Nc, int K, int act) {
    int tiles  = (M >> 4) * (Nc >> 5);      // 16x32 tiles per wave
    int blocks = (tiles + 3) / 4;           // 4 waves / 128-thread block
    switch (ldb) {
      case 512:
        hgt_wmma_gemm_bf16<512><<<blocks, 128, 0, stream>>>(A, lda, B, bias, C, ldc, M, Nc, K, act);
        break;
      case 256:
        hgt_wmma_gemm_bf16<256><<<blocks, 128, 0, stream>>>(A, lda, B, bias, C, ldc, M, Nc, K, act);
        break;
      case 128:
        hgt_wmma_gemm_bf16<128><<<blocks, 128, 0, stream>>>(A, lda, B, bias, C, ldc, M, Nc, K, act);
        break;
      default:
        hgt_wmma_gemm_bf16<64><<<blocks, 128, 0, stream>>>(A, lda, B, bias, C, ldc, M, Nc, K, act);
        break;
    }
  };

  const int ethreads = 256;                              // 8 waves / block
  const int eblocks  = (E + 7) / 8;
  const unsigned NEG_INF_KEY = 0x007FFFFFu;              // okey(-inf)

  auto hgt_layer = [&](const float* xin, const EncP& p, float* xout) {
    const int F = p.fout, D = F / H, fin = p.fin;
    gemm(xin, fin, p.Wk, F, p.bk, kbuf, F, N, F, fin, 0);
    gemm(xin, fin, p.Wq, F, p.bq, qbuf, F, N, F, fin, 0);
    gemm(xin, fin, p.Wv, F, p.bv, vbuf, F, N, F, fin, 0);
    for (int h = 0; h < H; ++h) {
      gemm(kbuf + h * D, F, p.a_rel + (size_t)h * D * D, D, nullptr,
           k2buf + h * D, F, N, D, D, 0);
      gemm(vbuf + h * D, F, p.m_rel + (size_t)h * D * D, D, nullptr,
           v2buf + h * D, F, N, D, D, 0);
    }
    hgt_edge_logits<<<eblocks, ethreads, 0, stream>>>(qbuf, k2buf, src, dst,
                                                      p.p_rel, alphaB, E, H, D);
    hgt_fill_u32<<<(N * H + 255) / 256, 256, 0, stream>>>(amaxk, NEG_INF_KEY, N * H);
    hgt_seg_max<<<(E * H + 255) / 256, 256, 0, stream>>>(alphaB, dst, amaxk, E, H);
    hgt_zero_f32<<<(N * H + 255) / 256, 256, 0, stream>>>(den, (long long)N * H);
    hgt_exp_den<<<(E * H + 255) / 256, 256, 0, stream>>>(alphaB, dst, amaxk, den, E, H);
    long long aggn = (long long)N * F;
    hgt_zero_f32<<<(int)((aggn + 255) / 256), 256, 0, stream>>>(aggbuf, aggn);
    hgt_scatter_v<<<eblocks, ethreads, 0, stream>>>(alphaB, v2buf, src, dst,
                                                    den, aggbuf, E, H, D);
    hgt_gelu<<<(int)((aggn + 255) / 256), 256, 0, stream>>>(aggbuf, aggn);
    gemm(aggbuf, F, p.Wa, F, p.ba, xout, F, N, F, F, 0);
  };

  // ---- branch O ----
  gemm(x, 512, lin_W, 512, lin_b, act_in, 512, N, 512, 512, /*relu*/1);
  hgt_layer(act_in, enc1, x1_o);
  hgt_layer(x1_o,   enc2, x2_o);

  // ---- branch SIM (reuse act_in, only N*256 needed) ----
  gemm(x_sim, 256, lin_sim_W, 256, lin_sim_b, act_in, 256, N, 256, 256, 1);
  hgt_layer(act_in, encs1, x1_s);
  hgt_layer(x1_s,   encs2, x2_s);

  // ---- readouts ----
  hgt_col_mean<<<128, 256, 0, stream>>>(x2_o, h_os, N, 128);
  hgt_col_mean<<<128, 256, 0, stream>>>(x2_s, h_os_sim, N, 128);
}
